// AggregationWithEpinions_8392366096484
// MI455X (gfx1250) — matrix operand order repaired and checked by reading, not compiled, not run on hardware
//
#include <hip/hip_runtime.h>
#include <hip/hip_bf16.h>

typedef __attribute__((ext_vector_type(16))) __bf16 v16bf;
typedef __attribute__((ext_vector_type(8)))  float  v8f;

#define NS   100000
#define ND   100000
#define NE   1600000
#define DD   64
#define NRAT 5
#define WAVES 8   // 8 wave32 per 256-thread block

__device__ __forceinline__ float lrelu(float x) { return x > 0.f ? x : 0.01f * x; }

__device__ __forceinline__ void zacc(v8f acc[4]) {
#pragma unroll
  for (int j = 0; j < 4; ++j)
#pragma unroll
    for (int r = 0; r < 8; ++r) acc[j][r] = 0.f;
}

// 16x64 (bf16, LDS row-major) @ 64x64 (bf16, fragment-packed) -> 4x v8f accumulators.
// A in contiguous k order (k = 16*laneHalf + s), as in k_pack; B fragments pre-packed so each lane loads
// 16 contiguous bf16 (32B) per (k,j) fragment.
__device__ __forceinline__ void wmma_tile(const __bf16* __restrict__ tb,
                                          const __bf16* __restrict__ pw,
                                          int lane, v8f acc[4]) {
  const int mrow = lane & 15;
  const int hi   = lane >> 4;
  const __bf16* row = tb + mrow * DD;
  const int b0 = hi * 16;       // lanes 0-15: K 0-7 ; lanes 16-31: K16-23
  const int b1 = 8 + hi * 16;   // lanes 0-15: K 8-15; lanes 16-31: K24-31
  v16bf a0, a1;
#pragma unroll
  for (int i = 0; i < 8; ++i) {
    a0[i]     = row[b0 + i];
    a0[i + 8] = row[b1 + i];
    a1[i]     = row[32 + b0 + i];
    a1[i + 8] = row[32 + b1 + i];
  }
#pragma unroll
  for (int j = 0; j < 4; ++j) {
    v16bf w0 = *(const v16bf*)(pw + ((size_t)((0 * 4 + j) * 32 + lane)) * 16);
    v16bf w1 = *(const v16bf*)(pw + ((size_t)((1 * 4 + j) * 32 + lane)) * 16);
    acc[j] = __builtin_amdgcn_wmma_f32_16x16x32_bf16(false, a0, false, w0,
                                                     (short)0, acc[j], false, false);
    acc[j] = __builtin_amdgcn_wmma_f32_16x16x32_bf16(false, a1, false, w1,
                                                     (short)0, acc[j], false, false);
  }
}

// Pack 7 weight matrices [64x64] f32 row-major -> bf16 B-fragment order.
// W[kk][n]: k=kk/32, i=kk%16, laneHalf=(kk%32)/16, j=n/16, lo=n%16, lane=lo+16*laneHalf
__global__ void k_pack(const float* __restrict__ w0, const float* __restrict__ w1,
                       const float* __restrict__ w2, const float* __restrict__ w3,
                       const float* __restrict__ w4, const float* __restrict__ w5,
                       const float* __restrict__ w6, __bf16* __restrict__ pk) {
  int t = blockIdx.x * blockDim.x + threadIdx.x;
  if (t >= 7 * DD * DD) return;
  int w = t >> 12, idx = t & 4095;
  const float* W;
  switch (w) {
    case 0: W = w0; break; case 1: W = w1; break; case 2: W = w2; break;
    case 3: W = w3; break; case 4: W = w4; break; case 5: W = w5; break;
    default: W = w6; break;
  }
  int kk = idx >> 6, n = idx & 63;
  int k = kk >> 5, i = kk & 15, lh = (kk >> 4) & 1;
  int j = n >> 4, lo = n & 15;
  int lane = lo + 16 * lh;
  pk[(size_t)w * 4096 + ((size_t)((k * 4 + j) * 32 + lane)) * 16 + i] = (__bf16)W[idx];
}

// fe_tab[5][64] = epinions @ W_ep + b_ep  (tiny; plain VALU)
__global__ void k_fetab(const float* __restrict__ ep, const float* __restrict__ W,
                        const float* __restrict__ b, float* __restrict__ out) {
  int t = blockIdx.x * blockDim.x + threadIdx.x;
  if (t >= NRAT * DD) return;
  int r = t / DD, n = t % DD;
  float acc = b[n];
#pragma unroll 8
  for (int k = 0; k < DD; ++k) acc += ep[r * DD + k] * W[k * DD + n];
  out[t] = acc;
}

// Generic node GEMM: Y = (lrelu?)(X @ W + b), one 16-row tile per wave.
__global__ void k_gemm(const float* __restrict__ X, const __bf16* __restrict__ pw,
                       const float* __restrict__ bias, float* __restrict__ Y,
                       int ntiles, int relu) {
  __shared__ alignas(16) __bf16 lbf[WAVES][16 * DD];
  const int lane = threadIdx.x & 31;
  const int wv   = threadIdx.x >> 5;
  const int tile = blockIdx.x * WAVES + wv;
  const bool act = tile < ntiles;
  __bf16* tb = lbf[wv];
  const int mrow2 = lane >> 1, half = lane & 1;
  if (act) {
    const size_t row = (size_t)tile * 16 + mrow2;
    const float4* px = (const float4*)(X + row * DD + half * 32);
    __bf16* d = tb + mrow2 * DD + half * 32;
#pragma unroll
    for (int q = 0; q < 8; ++q) {
      float4 a = px[q];
      d[4 * q + 0] = (__bf16)a.x; d[4 * q + 1] = (__bf16)a.y;
      d[4 * q + 2] = (__bf16)a.z; d[4 * q + 3] = (__bf16)a.w;
    }
  }
  __syncthreads();
  v8f acc[4]; zacc(acc);
  wmma_tile(tb, pw, lane, acc);
  const int lo = lane & 15, hi = lane >> 4;
  if (act) {
#pragma unroll
    for (int j = 0; j < 4; ++j) {
      float bj = bias[16 * j + lo];
#pragma unroll
      for (int r = 0; r < 8; ++r) {
        float v = acc[j][r] + bj;
        if (relu) v = lrelu(v);
        Y[((size_t)tile * 16 + r + 8 * hi) * DD + 16 * j + lo] = v;
      }
    }
  }
}

// Fused edge pipeline: 16 edges per wave, 3 chained WMMA GEMMs + per-edge score.
__global__ void k_edge(const float* __restrict__ fs, const float* __restrict__ rft,
                       const float* __restrict__ fetab,
                       const int* __restrict__ src_idx, const int* __restrict__ dst_idx,
                       const int* __restrict__ rating,
                       const __bf16* __restrict__ pw_comb, const __bf16* __restrict__ pw_ae,
                       const __bf16* __restrict__ pw_m1,
                       const float* __restrict__ b_comb, const float* __restrict__ b_ae,
                       const float* __restrict__ b_m1,
                       const float* __restrict__ w_m2, const float* __restrict__ b_m2,
                       __bf16* __restrict__ e_out, float* __restrict__ s_out, int ntiles) {
  __shared__ alignas(16) __bf16 lbf[WAVES][16 * DD];
  __shared__ alignas(16) float  lfl[WAVES][16 * DD];
  const int lane = threadIdx.x & 31;
  const int wv   = threadIdx.x >> 5;
  const int tile = blockIdx.x * WAVES + wv;
  const bool act = tile < ntiles;
  const int e0 = tile * 16;
  __bf16* tb = lbf[wv];
  float*  tf = lfl[wv];
  const int mrow2 = lane >> 1, half = lane & 1;
  const int lo = lane & 15, hi = lane >> 4;

  // Phase 1: stage h = lrelu(fs[src] + fe_tab[rating]) as bf16 tile
  if (act) {
    const int edge = e0 + mrow2;
    const float4* pf = (const float4*)(fs + (size_t)src_idx[edge] * DD + half * 32);
    const float4* pe = (const float4*)(fetab + (size_t)rating[edge] * DD + half * 32);
    __bf16* d = tb + mrow2 * DD + half * 32;
#pragma unroll
    for (int q = 0; q < 8; ++q) {
      float4 a = pf[q], b = pe[q];
      d[4 * q + 0] = (__bf16)lrelu(a.x + b.x);
      d[4 * q + 1] = (__bf16)lrelu(a.y + b.y);
      d[4 * q + 2] = (__bf16)lrelu(a.z + b.z);
      d[4 * q + 3] = (__bf16)lrelu(a.w + b.w);
    }
  }
  __syncthreads();

  // GEMM1: e = lrelu(h @ W_comb + b_comb)
  v8f acc[4]; zacc(acc);
  wmma_tile(tb, pw_comb, lane, acc);
  __syncthreads();
#pragma unroll
  for (int j = 0; j < 4; ++j) {
    float bj = b_comb[16 * j + lo];
#pragma unroll
    for (int r = 0; r < 8; ++r)
      tb[(r + 8 * hi) * DD + 16 * j + lo] = (__bf16)lrelu(acc[j][r] + bj);
  }
  __syncthreads();

  // Store e tile (bf16) for the aggregation pass
  if (act) {
    const int edge = e0 + mrow2;
    const uint4* sp = (const uint4*)(tb + mrow2 * DD + half * 32);
    uint4* dp = (uint4*)(e_out + (size_t)edge * DD + half * 32);
#pragma unroll
    for (int q = 0; q < 4; ++q) dp[q] = sp[q];
  }

  // GEMM2: e @ W_ae ; stage r_ft[dst] tile (f32) concurrently
  zacc(acc);
  wmma_tile(tb, pw_ae, lane, acc);
  if (act) {
    const int edge = e0 + mrow2;
    const float4* pr = (const float4*)(rft + (size_t)dst_idx[edge] * DD + half * 32);
    float4* d = (float4*)(tf + mrow2 * DD + half * 32);
#pragma unroll
    for (int q = 0; q < 8; ++q) d[q] = pr[q];
  }
  __syncthreads();

  // u = lrelu(acc + b_ae + r_ft) -> tb (bf16)
#pragma unroll
  for (int j = 0; j < 4; ++j) {
    float bj = b_ae[16 * j + lo];
#pragma unroll
    for (int r = 0; r < 8; ++r) {
      float v = lrelu(acc[j][r] + bj + tf[(r + 8 * hi) * DD + 16 * j + lo]);
      tb[(r + 8 * hi) * DD + 16 * j + lo] = (__bf16)v;
    }
  }
  __syncthreads();

  // GEMM3: v = lrelu(u @ W_m1 + b_m1) -> tf (f32)
  zacc(acc);
  wmma_tile(tb, pw_m1, lane, acc);
  __syncthreads();
#pragma unroll
  for (int j = 0; j < 4; ++j) {
    float bj = b_m1[16 * j + lo];
#pragma unroll
    for (int r = 0; r < 8; ++r)
      tf[(r + 8 * hi) * DD + 16 * j + lo] = lrelu(acc[j][r] + bj);
  }
  __syncthreads();

  // s = v . W_m2 + b_m2  (one edge per lane 0..15)
  if (act && lane < 16) {
    const float* vr = tf + lane * DD;
    float s = b_m2[0];
#pragma unroll 8
    for (int d2 = 0; d2 < DD; ++d2) s += vr[d2] * w_m2[d2];
    s_out[e0 + lane] = s;
  }
}

__global__ void k_init(float* __restrict__ m, float* __restrict__ den,
                       float* __restrict__ nf, int nd) {
  int t = blockIdx.x * blockDim.x + threadIdx.x;
  if (t < nd * DD) nf[t] = 0.f;
  if (t < nd) { m[t] = -__builtin_inff(); den[t] = 0.f; }
}

// segment max via monotonic int-punning atomics
__global__ void k_max(const float* __restrict__ s, const int* __restrict__ dst,
                      float* __restrict__ m, int ne) {
  int t = blockIdx.x * blockDim.x + threadIdx.x;
  if (t >= ne) return;
  float v = s[t];
  int d = dst[t];
  if (v >= 0.f) atomicMax((int*)(m + d), __float_as_int(v));
  else          atomicMin((unsigned int*)(m + d), (unsigned int)__float_as_int(v));
}

__global__ void k_exp(float* __restrict__ s, const int* __restrict__ dst,
                      const float* __restrict__ m, float* __restrict__ den, int ne) {
  int t = blockIdx.x * blockDim.x + threadIdx.x;
  if (t >= ne) return;
  int d = dst[t];
  float ex = __expf(s[t] - m[d]);
  s[t] = ex;                       // s becomes ex (in-place, deterministic)
  atomicAdd(den + d, ex);
}

__global__ void k_agg(const float* __restrict__ ex, const int* __restrict__ dst,
                      const float* __restrict__ den, const __bf16* __restrict__ ebf,
                      float* __restrict__ nf, int ne) {
  int t = blockIdx.x * blockDim.x + threadIdx.x;
  if (t >= ne * DD) return;
  int e = t >> 6, d = t & 63;
  int dn = dst[e];
  float alpha = ex[e] / den[dn];
  atomicAdd(nf + (size_t)dn * DD + d, alpha * (float)ebf[(size_t)e * DD + d]);
}

extern "C" void kernel_launch(void* const* d_in, const int* in_sizes, int n_in,
                              void* d_out, int out_size, void* d_ws, size_t ws_size,
                              hipStream_t stream) {
  (void)in_sizes; (void)n_in; (void)out_size; (void)ws_size;
  const float* feat_src = (const float*)d_in[0];
  const float* feat_dst = (const float*)d_in[1];
  const float* epinions = (const float*)d_in[2];
  const int*   src_idx  = (const int*)d_in[3];
  const int*   dst_idx  = (const int*)d_in[4];
  const int*   rating   = (const int*)d_in[5];
  const float* W_src = (const float*)d_in[6];  const float* b_src = (const float*)d_in[7];
  const float* W_dst = (const float*)d_in[8];  const float* b_dst = (const float*)d_in[9];
  const float* W_ep  = (const float*)d_in[10]; const float* b_ep  = (const float*)d_in[11];
  const float* W_comb= (const float*)d_in[12]; const float* b_comb= (const float*)d_in[13];
  const float* W_ae  = (const float*)d_in[14]; const float* b_ae  = (const float*)d_in[15];
  const float* W_ad  = (const float*)d_in[16]; const float* b_ad  = (const float*)d_in[17];
  const float* W_m1  = (const float*)d_in[18]; const float* b_m1  = (const float*)d_in[19];
  const float* W_m2  = (const float*)d_in[20]; const float* b_m2  = (const float*)d_in[21];
  const float* W_out = (const float*)d_in[22]; const float* b_out = (const float*)d_in[23];

  // workspace layout (~289 MB total)
  char* ws = (char*)d_ws;
  size_t off = 0;
  float* fsb   = (float*)(ws + off); off += (size_t)NS * DD * 4;    // fs
  float* rftb  = (float*)(ws + off); off += (size_t)ND * DD * 4;    // r_ft
  float* nfb   = (float*)(ws + off); off += (size_t)ND * DD * 4;    // nf (also fd tmp)
  float* fetab = (float*)(ws + off); off += (size_t)NRAT * DD * 4;  // fe table
  off = (off + 255) & ~(size_t)255;
  __bf16* packw = (__bf16*)(ws + off); off += (size_t)7 * DD * DD * 2;
  off = (off + 255) & ~(size_t)255;
  float* mbuf = (float*)(ws + off); off += (size_t)ND * 4;
  float* denb = (float*)(ws + off); off += (size_t)ND * 4;
  float* sbuf = (float*)(ws + off); off += (size_t)NE * 4;
  off = (off + 255) & ~(size_t)255;
  __bf16* ebf = (__bf16*)(ws + off); off += (size_t)NE * DD * 2;

  // pack order: 0=W_src 1=W_dst 2=W_ad 3=W_comb 4=W_ae 5=W_m1 6=W_out
  k_pack<<<(7 * DD * DD + 255) / 256, 256, 0, stream>>>(W_src, W_dst, W_ad, W_comb,
                                                        W_ae, W_m1, W_out, packw);
  k_fetab<<<2, 256, 0, stream>>>(epinions, W_ep, b_ep, fetab);

  const int ntN  = NS / 16;                     // 6250 tiles (exact)
  const int nbN  = (ntN + WAVES - 1) / WAVES;
  k_gemm<<<nbN, 256, 0, stream>>>(feat_src, packw + 0 * 4096, b_src, fsb,  ntN, 0);
  k_gemm<<<nbN, 256, 0, stream>>>(feat_dst, packw + 1 * 4096, b_dst, nfb,  ND / 16, 0); // fd -> tmp
  k_gemm<<<nbN, 256, 0, stream>>>(nfb,      packw + 2 * 4096, b_ad,  rftb, ND / 16, 0); // r_ft

  const int ntE = NE / 16;                      // 100000 tiles (exact)
  k_edge<<<(ntE + WAVES - 1) / WAVES, 256, 0, stream>>>(
      fsb, rftb, fetab, src_idx, dst_idx, rating,
      packw + 3 * 4096, packw + 4 * 4096, packw + 5 * 4096,
      b_comb, b_ae, b_m1, W_m2, b_m2, ebf, sbuf, ntE);

  k_init<<<(ND * DD + 255) / 256, 256, 0, stream>>>(mbuf, denb, nfb, ND);
  k_max<<<(NE + 255) / 256, 256, 0, stream>>>(sbuf, dst_idx, mbuf, NE);
  k_exp<<<(NE + 255) / 256, 256, 0, stream>>>(sbuf, dst_idx, mbuf, denb, NE);
  k_agg<<<(NE * DD + 255) / 256, 256, 0, stream>>>(sbuf, dst_idx, denb, ebf, nfb, NE);

  k_gemm<<<nbN, 256, 0, stream>>>(nfb, packw + 6 * 4096, b_out, (float*)d_out, ND / 16, 1);
}
